// AttentionBasedPooling_50912542327318
// MI455X (gfx1250) — matrix operand order repaired and checked by reading, not compiled, not run on hardware
//
#include <hip/hip_runtime.h>

typedef __attribute__((ext_vector_type(16))) _Float16 v16h;
typedef __attribute__((ext_vector_type(8)))  float    v8f;

#define N_TOK 1048576
#define BATCH 4096
#define DIM   256
#define HID   128

// ---------------- kernel 0: init outputs / scratch ----------------
__global__ void k_init(float* __restrict__ ctx, float* __restrict__ segmax,
                       float* __restrict__ segsum) {
    int i = blockIdx.x * blockDim.x + threadIdx.x;
    if (i < BATCH * DIM) ctx[i] = 0.0f;
    if (i < BATCH) { segmax[i] = -INFINITY; segsum[i] = 0.0f; }
}

// ---------------- kernel 1: pack W1 into WMMA B-fragment image (f16) ------
// Layout: pw1[ nt*4096 + kt*512 + lane*16 + e ] = W1[kt*32 + (lane/16)*8 + (e&7) + (e>>3)*16][nt*16 + lane%16]
// Matches ISA 16-bit B 32x16 layout: lanes 0-15 K=0..7/16..23, lanes 16-31 K=8..15/24..31.
__global__ void k_pack_w1(const float* __restrict__ W1, _Float16* __restrict__ pw1) {
    int tid = blockIdx.x * blockDim.x + threadIdx.x;   // 0 .. 32767
    int e    = tid & 15;
    int lane = (tid >> 4) & 31;
    int kt   = (tid >> 9) & 7;
    int nt   = tid >> 12;
    int m  = lane & 15;
    int hf = lane >> 4;
    int k = kt * 32 + hf * 8 + (e & 7) + (e >> 3) * 16;
    int n = nt * 16 + m;
    pw1[tid] = (_Float16)W1[k * HID + n];
}

// ---------------- kernel 2: logits via WMMA -------------------------------
// s[i] = tanh(x[i] @ W1 + b1) @ W2 + b2
__global__ void __launch_bounds__(256)
k_logits(const float* __restrict__ x, const _Float16* __restrict__ pw1,
         const float* __restrict__ b1, const float* __restrict__ W2,
         const float* __restrict__ b2, float* __restrict__ s) {
    const int lane = threadIdx.x & 31;
    const int wave = threadIdx.x >> 5;
    const int m  = lane & 15;
    const int hf = lane >> 4;                 // 0: K 0..7/16..23, 1: K 8..15/24..31
    const int rowBase = blockIdx.x * 128 + wave * 16;
    const int rowA = rowBase + m;

    float b1v[8], w2v[8];
#pragma unroll
    for (int nt = 0; nt < 8; ++nt) { b1v[nt] = b1[nt * 16 + m]; w2v[nt] = W2[nt * 16 + m]; }

    // Load all A fragments for this wave's 16 rows (K = 0..255), x read once.
    v16h Afr[8];
#pragma unroll
    for (int kt = 0; kt < 8; ++kt) {
        const float* xr = x + (size_t)rowA * DIM + kt * 32 + hf * 8;
        float4 a0 = *(const float4*)(xr);
        float4 a1 = *(const float4*)(xr + 4);
        float4 a2 = *(const float4*)(xr + 16);
        float4 a3 = *(const float4*)(xr + 20);
        v16h A;
        A[0]=(_Float16)a0.x; A[1]=(_Float16)a0.y; A[2]=(_Float16)a0.z; A[3]=(_Float16)a0.w;
        A[4]=(_Float16)a1.x; A[5]=(_Float16)a1.y; A[6]=(_Float16)a1.z; A[7]=(_Float16)a1.w;
        A[8]=(_Float16)a2.x; A[9]=(_Float16)a2.y; A[10]=(_Float16)a2.z; A[11]=(_Float16)a2.w;
        A[12]=(_Float16)a3.x; A[13]=(_Float16)a3.y; A[14]=(_Float16)a3.z; A[15]=(_Float16)a3.w;
        Afr[kt] = A;
    }

    float partial[8];
#pragma unroll
    for (int v = 0; v < 8; ++v) partial[v] = 0.0f;

#pragma unroll
    for (int nt = 0; nt < 8; ++nt) {
        v8f acc = {};
#pragma unroll
        for (int kt = 0; kt < 8; ++kt) {
            v16h Bf = *(const v16h*)(pw1 + ((nt * 8 + kt) * 32 + lane) * 16);
            acc = __builtin_amdgcn_wmma_f32_16x16x32_f16(
                false, Afr[kt], false, Bf, (short)0, acc, false, false);
        }
        // acc[v] = h[row = rowBase + v + 8*hf][col = nt*16 + m]
#pragma unroll
        for (int v = 0; v < 8; ++v) {
            float t = tanhf(acc[v] + b1v[nt]);
            partial[v] += t * w2v[nt];
        }
    }

    // Reduce over the 16 column-lanes inside each half (xor bits 0..3 stay in-half)
#pragma unroll
    for (int off = 1; off < 16; off <<= 1)
#pragma unroll
        for (int v = 0; v < 8; ++v)
            partial[v] += __shfl_xor(partial[v], off, 32);

    if (m == 0) {
        float bb = b2[0];
#pragma unroll
        for (int v = 0; v < 8; ++v)
            s[rowBase + hf * 8 + v] = partial[v] + bb;
    }
}

// ---------------- kernel 3: per-segment max (sorted ids -> wave fast path) -
__device__ inline void atomicMaxF(float* a, float v) {
    if (v >= 0.0f) atomicMax((int*)a, __float_as_int(v));
    else           atomicMin((unsigned int*)a, __float_as_uint(v));
}

__global__ void k_segmax(const float* __restrict__ s, const int* __restrict__ seg,
                         float* __restrict__ segmax) {
    int i = blockIdx.x * blockDim.x + threadIdx.x;
    int sg  = seg[i];
    float v = s[i];
    int sg0 = __shfl(sg, 0, 32);
    if (__all(sg == sg0)) {
        for (int off = 16; off; off >>= 1) v = fmaxf(v, __shfl_xor(v, off, 32));
        if ((threadIdx.x & 31) == 0) atomicMaxF(&segmax[sg0], v);
    } else {
        atomicMaxF(&segmax[sg], v);
    }
}

// ---------------- kernel 4: z = exp(s - max), segment sum ------------------
__global__ void k_expsum(float* __restrict__ sz, const int* __restrict__ seg,
                         const float* __restrict__ segmax, float* __restrict__ segsum) {
    int i = blockIdx.x * blockDim.x + threadIdx.x;
    int sg  = seg[i];
    float z = expf(sz[i] - segmax[sg]);
    sz[i] = z;                                 // s buffer now holds z
    int sg0 = __shfl(sg, 0, 32);
    if (__all(sg == sg0)) {
        float t = z;
        for (int off = 16; off; off >>= 1) t += __shfl_xor(t, off, 32);
        if ((threadIdx.x & 31) == 0) atomicAdd(&segsum[sg0], t);
    } else {
        atomicAdd(&segsum[sg], z);
    }
}

// ---------------- kernel 5: weights + segment-weighted pooling -------------
// Block = 256 rows, thread t owns column t; register accumulator flushed
// to global ctx only when the (sorted) segment id changes.
__global__ void __launch_bounds__(256)
k_pool(const float* __restrict__ x, const int* __restrict__ seg,
       const float* __restrict__ z, const float* __restrict__ segsum,
       float* __restrict__ ctx, float* __restrict__ wout) {
    const int t  = threadIdx.x;
    const int i0 = blockIdx.x * 256;

    {   // coalesced attention-weight store
        int iw = i0 + t;
        wout[iw] = z[iw] / segsum[seg[iw]];
    }

    float acc = 0.0f;
    int cur = seg[i0];
    for (int i = i0; i < i0 + 256; ++i) {
        int sg = seg[i];
        if (sg != cur) {
            atomicAdd(&ctx[cur * DIM + t], acc);
            acc = 0.0f;
            cur = sg;
        }
        float w = z[i] / segsum[sg];
        acc = fmaf(w, x[(size_t)i * DIM + t], acc);
    }
    atomicAdd(&ctx[cur * DIM + t], acc);
}

// ---------------- launcher -------------------------------------------------
extern "C" void kernel_launch(void* const* d_in, const int* in_sizes, int n_in,
                              void* d_out, int out_size, void* d_ws, size_t ws_size,
                              hipStream_t stream) {
    (void)in_sizes; (void)n_in; (void)out_size; (void)ws_size;
    const float* x   = (const float*)d_in[0];
    const int*   seg = (const int*)d_in[1];
    const float* W1  = (const float*)d_in[2];
    const float* b1  = (const float*)d_in[3];
    const float* W2  = (const float*)d_in[4];
    const float* b2  = (const float*)d_in[5];

    float* out  = (float*)d_out;
    float* ctx  = out;                  // [BATCH*DIM]
    float* wout = out + BATCH * DIM;    // [N_TOK]

    float* ws_f   = (float*)d_ws;
    float* s      = ws_f;               // [N_TOK] logits, then z
    float* segmax = ws_f + N_TOK;       // [BATCH]
    float* segsum = segmax + BATCH;     // [BATCH]
    _Float16* pw1 = (_Float16*)(segsum + BATCH);  // [32768] packed W1

    k_init   <<<BATCH * DIM / 256, 256, 0, stream>>>(ctx, segmax, segsum);
    k_pack_w1<<<128,               256, 0, stream>>>(W1, pw1);
    k_logits <<<N_TOK / 128,       256, 0, stream>>>(x, pw1, b1, W2, b2, s);
    k_segmax <<<N_TOK / 256,       256, 0, stream>>>(s, seg, segmax);
    k_expsum <<<N_TOK / 256,       256, 0, stream>>>(s, seg, segmax, segsum);
    k_pool   <<<N_TOK / 256,       256, 0, stream>>>(x, seg, s, segsum, ctx, wout);
}
